// LinkPredictor_48172353192035
// MI455X (gfx1250) — matrix operand order repaired and checked by reading, not compiled, not run on hardware
//
#include <hip/hip_runtime.h>

#define N_NODES 100000
#define N_EDGES 1200000
#define IN_CH 16
#define HID 64

typedef __attribute__((ext_vector_type(16))) __bf16 v16bf;
typedef __attribute__((ext_vector_type(8)))  float  v8f;

// Convert two float4s into 8 consecutive bf16 lanes of an A fragment.
__device__ __forceinline__ void cvt8(v16bf& a, int base, float4 f0, float4 f1, float s) {
    a[base + 0] = (__bf16)(f0.x * s);
    a[base + 1] = (__bf16)(f0.y * s);
    a[base + 2] = (__bf16)(f0.z * s);
    a[base + 3] = (__bf16)(f0.w * s);
    a[base + 4] = (__bf16)(f1.x * s);
    a[base + 5] = (__bf16)(f1.y * s);
    a[base + 6] = (__bf16)(f1.z * s);
    a[base + 7] = (__bf16)(f1.w * s);
}

// Build the 16x32 bf16 A fragment for one lane from a single row's 32-float
// K-window.  ISA layout: elem i<8 -> K = half*8+i ; elem i>=8 -> K = 16+half*8+(i-8).
__device__ __forceinline__ v16bf load_a32(const float* __restrict__ row, int half, float s) {
    const float4* p = (const float4*)(row + half * 8);
    const float4* q = (const float4*)(row + 16 + half * 8);
    v16bf a;
    cvt8(a, 0, p[0], p[1], s);
    cvt8(a, 8, q[0], q[1], s);
    return a;
}

// ---------------------------------------------------------------- zero scratch
__global__ void zero_kernel(float4* __restrict__ p, long n4) {
    long i = (long)blockIdx.x * blockDim.x + threadIdx.x;
    long stride = (long)gridDim.x * blockDim.x;
    float4 z = {0.f, 0.f, 0.f, 0.f};
    for (; i < n4; i += stride) p[i] = z;
}

// ------------------------------------------------- layer-1 scatter (16 ch/edge)
__global__ void scatter1_kernel(const float* __restrict__ x,
                                const long long* __restrict__ ei,
                                float* __restrict__ agg1,
                                float* __restrict__ deg) {
    long t = (long)blockIdx.x * blockDim.x + threadIdx.x;
    if (t >= (long)N_EDGES * IN_CH) return;
    int e = (int)(t >> 4);
    int c = (int)(t & 15);
    int s = (int)ei[e];
    int d = (int)ei[N_EDGES + e];
    atomicAdd(&agg1[(long)d * IN_CH + c], x[(long)s * IN_CH + c]);
    if (c == 0) atomicAdd(&deg[d], 1.0f);
}

// ------------------------------------------------- layer-2 scatter (64 ch/edge)
__global__ void scatter2_kernel(const float* __restrict__ h1,
                                const long long* __restrict__ ei,
                                float* __restrict__ agg2) {
    long t = (long)blockIdx.x * blockDim.x + threadIdx.x;
    if (t >= (long)N_EDGES * HID) return;
    int e = (int)(t >> 6);
    int c = (int)(t & 63);
    int s = (int)ei[e];
    int d = (int)ei[N_EDGES + e];
    atomicAdd(&agg2[(long)d * HID + c], h1[(long)s * HID + c]);
}

// ---------------- SAGE layer 1: h1 = relu([agg1/deg | x] * [W1l;W1r]^T + b1l)
// K = 32, N = 64.  One wave per 16-node tile, 8 waves/block.
// B staged in LDS pre-swizzled to fragment order: [nt][lane][16 bf16].
__global__ void __launch_bounds__(256) sage1_gemm_kernel(
    const float* __restrict__ x,   const float* __restrict__ agg1,
    const float* __restrict__ deg,
    const float* __restrict__ W1l, const float* __restrict__ b1l,
    const float* __restrict__ W1r, float* __restrict__ h1) {
    __shared__ __attribute__((aligned(32))) __bf16 Bs[4 * 32 * 16];
    for (int idx = threadIdx.x; idx < 4 * 32 * 16; idx += 256) {
        int i = idx & 15;
        int ln = (idx >> 4) & 31;
        int nt = idx >> 9;
        int k = (i >> 3) * 16 + (ln >> 4) * 8 + (i & 7);   // B elem i<8 -> K = half*8+i ; i>=8 -> K = 16+half*8+(i-8)
        int n = nt * 16 + (ln & 15);
        float w = (k < 16) ? W1l[n * 16 + k] : W1r[n * 16 + (k - 16)];
        Bs[idx] = (__bf16)w;
    }
    __syncthreads();

    int wave = threadIdx.x >> 5;
    int tile = blockIdx.x * 8 + wave;
    if (tile >= N_NODES / 16) return;
    int lane = threadIdx.x & 31;
    int m = lane & 15, half = lane >> 4;
    int node = tile * 16 + m;
    float dinv = 1.0f / fmaxf(deg[node], 1.0f);

    // A: K<16 from agg1/deg, K>=16 from x (two contiguous float4 pairs).
    v16bf a;
    {
        const float4* p = (const float4*)(agg1 + (long)node * 16 + half * 8);
        cvt8(a, 0, p[0], p[1], dinv);
        const float4* q = (const float4*)(x + (long)node * 16 + half * 8);
        cvt8(a, 8, q[0], q[1], 1.0f);
    }

    const v16bf* Bfrag = (const v16bf*)Bs;
    v8f acc[4];
#pragma unroll
    for (int nt = 0; nt < 4; ++nt) {
        v16bf b = Bfrag[nt * 32 + lane];     // 2x ds_load_b128
        v8f c = {};
        acc[nt] = __builtin_amdgcn_wmma_f32_16x16x32_bf16(
            false, a, false, b, (short)0, c, false, false);
    }

#pragma unroll
    for (int nt = 0; nt < 4; ++nt) {
        int n = nt * 16 + m;
        float bias = b1l[n];
#pragma unroll
        for (int r = 0; r < 8; ++r) {
            int row = tile * 16 + r + 8 * half;
            h1[(long)row * HID + n] = fmaxf(acc[nt][r] + bias, 0.0f);
        }
    }
}

// ---------------- SAGE layer 2: h2 = [agg2/deg | h1] * [W2l;W2r]^T + b2l
// K = 128 (4 steps of 32), N = 64.  LDS B fragments: [step][nt][lane][16].
__global__ void __launch_bounds__(256) sage2_gemm_kernel(
    const float* __restrict__ h1,  const float* __restrict__ agg2,
    const float* __restrict__ deg,
    const float* __restrict__ W2l, const float* __restrict__ b2l,
    const float* __restrict__ W2r, float* __restrict__ h2) {
    __shared__ __attribute__((aligned(32))) __bf16 Bs[16 * 32 * 16];   // 16 KB
    for (int idx = threadIdx.x; idx < 16 * 32 * 16; idx += 256) {
        int i = idx & 15;
        int ln = (idx >> 4) & 31;
        int nt = (idx >> 9) & 3;
        int s  = idx >> 11;
        int k = s * 32 + (i >> 3) * 16 + (ln >> 4) * 8 + (i & 7);
        int n = nt * 16 + (ln & 15);
        float w = (k < 64) ? W2l[n * 64 + k] : W2r[n * 64 + (k - 64)];
        Bs[idx] = (__bf16)w;
    }
    __syncthreads();

    int wave = threadIdx.x >> 5;
    int tile = blockIdx.x * 8 + wave;
    if (tile >= N_NODES / 16) return;
    int lane = threadIdx.x & 31;
    int m = lane & 15, half = lane >> 4;
    int node = tile * 16 + m;
    float dinv = 1.0f / fmaxf(deg[node], 1.0f);
    const float* arow = agg2 + (long)node * HID;
    const float* hrow = h1 + (long)node * HID;
    const v16bf* Bfrag = (const v16bf*)Bs;

    v8f acc[4] = {};
#pragma unroll
    for (int s = 0; s < 4; ++s) {
        v16bf a = (s < 2) ? load_a32(arow + s * 32, half, dinv)
                          : load_a32(hrow + (s - 2) * 32, half, 1.0f);
#pragma unroll
        for (int nt = 0; nt < 4; ++nt) {
            v16bf b = Bfrag[(s * 4 + nt) * 32 + lane];
            acc[nt] = __builtin_amdgcn_wmma_f32_16x16x32_bf16(
                false, a, false, b, (short)0, acc[nt], false, false);
        }
    }

#pragma unroll
    for (int nt = 0; nt < 4; ++nt) {
        int n = nt * 16 + m;
        float bias = b2l[n];
#pragma unroll
        for (int r = 0; r < 8; ++r) {
            int row = tile * 16 + r + 8 * half;
            h2[(long)row * HID + n] = acc[nt][r] + bias;   // no activation
        }
    }
}

// ---------------- Edge MLP: out[e] = relu([h2[s]|h2[d]]·Wm1^T + bm1)·Wm2^T + bm2
// K = 128 (4 steps), N = 16; 16 edges per wave.  LDS B fragments: [step][lane][16].
__global__ void __launch_bounds__(256) mlp_edge_kernel(
    const float* __restrict__ h2, const long long* __restrict__ ei,
    const float* __restrict__ Wm1, const float* __restrict__ bm1,
    const float* __restrict__ Wm2, const float* __restrict__ bm2,
    float* __restrict__ out) {
    __shared__ __attribute__((aligned(32))) __bf16 Bs[4 * 32 * 16];
    for (int idx = threadIdx.x; idx < 4 * 32 * 16; idx += 256) {
        int i = idx & 15;
        int ln = (idx >> 4) & 31;
        int st = idx >> 9;
        int k = st * 32 + (i >> 3) * 16 + (ln >> 4) * 8 + (i & 7);
        int n = ln & 15;
        Bs[idx] = (__bf16)Wm1[n * 128 + k];
    }
    __syncthreads();

    int wave = threadIdx.x >> 5;
    int tile = blockIdx.x * 8 + wave;
    if (tile >= N_EDGES / 16) return;
    int lane = threadIdx.x & 31;
    int m = lane & 15, half = lane >> 4;
    int e = tile * 16 + m;
    long s = (long)ei[e];
    long d = (long)ei[N_EDGES + e];
    const float* srow = h2 + s * HID;
    const float* drow = h2 + d * HID;
    const v16bf* Bfrag = (const v16bf*)Bs;

    v8f acc = {};
#pragma unroll
    for (int st = 0; st < 4; ++st) {
        v16bf a = (st < 2) ? load_a32(srow + st * 32, half, 1.0f)
                           : load_a32(drow + (st - 2) * 32, half, 1.0f);
        v16bf b = Bfrag[st * 32 + lane];
        acc = __builtin_amdgcn_wmma_f32_16x16x32_bf16(
            false, a, false, b, (short)0, acc, false, false);
    }

    // second MLP layer: relu + weighted reduce across the 16 hidden channels
    float bias = bm1[m];
    float w2 = Wm2[m];
    float partial[8];
#pragma unroll
    for (int r = 0; r < 8; ++r)
        partial[r] = fmaxf(acc[r] + bias, 0.0f) * w2;
#pragma unroll
    for (int mask = 1; mask < 16; mask <<= 1) {
#pragma unroll
        for (int r = 0; r < 8; ++r)
            partial[r] += __shfl_xor(partial[r], mask, 32);
    }
    if (m == 0) {
        float b2 = bm2[0];
#pragma unroll
        for (int r = 0; r < 8; ++r)
            out[tile * 16 + r + 8 * half] = partial[r] + b2;
    }
}

extern "C" void kernel_launch(void* const* d_in, const int* in_sizes, int n_in,
                              void* d_out, int out_size, void* d_ws, size_t ws_size,
                              hipStream_t stream) {
    const float*     x   = (const float*)d_in[0];
    const long long* ei  = (const long long*)d_in[1];   // int64 edge_index [2,E]
    const float*     W1l = (const float*)d_in[2];
    const float*     b1l = (const float*)d_in[3];
    const float*     W1r = (const float*)d_in[4];
    const float*     W2l = (const float*)d_in[5];
    const float*     b2l = (const float*)d_in[6];
    const float*     W2r = (const float*)d_in[7];
    const float*     Wm1 = (const float*)d_in[8];
    const float*     bm1 = (const float*)d_in[9];
    const float*     Wm2 = (const float*)d_in[10];
    const float*     bm2 = (const float*)d_in[11];
    float* out = (float*)d_out;

    // workspace layout (floats): deg | agg1 | agg2 | h1 | h2  (209*N ~ 84 MB)
    float* ws   = (float*)d_ws;
    float* deg  = ws;
    float* agg1 = ws + (long)N_NODES;
    float* agg2 = ws + (long)N_NODES * 17;
    float* h1   = ws + (long)N_NODES * 81;
    float* h2   = ws + (long)N_NODES * 145;

    // 1) zero deg + agg1 + agg2  (81*N floats, divisible by 4)
    zero_kernel<<<2048, 256, 0, stream>>>((float4*)ws, (long)N_NODES * 81 / 4);

    // 2) layer-1 mean-agg scatter + degree
    {
        long t = (long)N_EDGES * IN_CH;
        scatter1_kernel<<<(int)((t + 255) / 256), 256, 0, stream>>>(x, ei, agg1, deg);
    }
    // 3) layer-1 dense transform (WMMA) + ReLU
    sage1_gemm_kernel<<<(N_NODES / 16 + 7) / 8, 256, 0, stream>>>(
        x, agg1, deg, W1l, b1l, W1r, h1);

    // 4) layer-2 scatter
    {
        long t = (long)N_EDGES * HID;
        scatter2_kernel<<<(int)((t + 255) / 256), 256, 0, stream>>>(h1, ei, agg2);
    }
    // 5) layer-2 dense transform (WMMA), no activation
    sage2_gemm_kernel<<<(N_NODES / 16 + 7) / 8, 256, 0, stream>>>(
        h1, agg2, deg, W2l, b2l, W2r, h2);

    // 6) per-edge MLP (WMMA) -> scalar score
    mlp_edge_kernel<<<(N_EDGES / 16 + 7) / 8, 256, 0, stream>>>(
        h2, ei, Wm1, bm1, Wm2, bm2, out);
}
